// TOFBranch_39719857553893
// MI455X (gfx1250) — matrix operand order repaired and checked by reading, not compiled, not run on hardware
//
#include <hip/hip_runtime.h>
#include <hip/hip_bf16.h>
#include <cstdint>
#include <cstddef>

// ---------------------------------------------------------------------------
// TOFBranch forward for MI455X (gfx1250, wave32).
// Core: f16 WMMA GEMM (v_wmma_f32_16x16x32_f16) with fused BN/SiLU/ReLU/
// sigmoid/SE-multiply/residual epilogues. Blocks 4-15 degenerate to pure
// per-row matmuls (1x1 spatial), which dominate (~0.3 TFLOP over 20480 rows).
// Workspace layout (assumes ws_size >= ~200 MB):
//   [0,32MB)    packed f16 weights (fragment order)
//   [32,96MB)   bufY  (expand/dw/head activations, max 63MB)
//   [96,120MB)  bufX  (block input / residual)
//   [120,144MB) bufZ  (project output; final f32 features)
//   [144,152MB) bufP  (SE pooled / sigmoid y, multi-pixel blocks)
//   [152,164MB) bufS1 (SE hidden)
//   [164,196MB) bufD  (dw output, multi-pixel blocks)
// ---------------------------------------------------------------------------

typedef __attribute__((ext_vector_type(16))) _Float16 v16h;
typedef __attribute__((ext_vector_type(8)))  _Float16 v8h;
typedef __attribute__((ext_vector_type(8)))  float    v8f;

#define BNRS 0.9999950000374997f  /* 1/sqrt(1+1e-5) */
#define NIMG 20480

enum { EPI_NONE = 0, EPI_SILU = 1, EPI_RELU = 2, EPI_SIG = 3, EPI_MULSIG = 4 };

static inline int rup(int x, int m) { return (x + m - 1) / m * m; }

// ------------------------- weight pre-pack (f32 -> f16 fragments) ----------
// B-matrix fragment order for V_WMMA_*_16X16X32: lane L holds column n=L%16,
// k = 8*(L/16)+j for j<8, k = 16+8*(L/16)+(j-8) for j>=8, for each (ntile, ktile).
__global__ __launch_bounds__(256) void pack_w(const float* __restrict__ W,
                                              _Float16* __restrict__ P,
                                              int K, int N, int KT, long total) {
  long idx = (long)blockIdx.x * 256 + threadIdx.x;
  if (idx >= total) return;
  int j = (int)(idx & 15);
  int lane = (int)((idx >> 4) & 31);
  long t = idx >> 9;
  int kt = (int)(t % KT);
  long nt = t / KT;
  int n = (int)(nt * 16 + (lane & 15));
  int k = kt * 32 + (lane >> 4) * 8 + (j & 7) + (j >> 3) * 16;
  float v = (n < N && k < K) ? W[(size_t)n * K + k] : 0.0f;
  P[idx] = (_Float16)v;
}

// ------------------------- WMMA GEMM with fused epilogue -------------------
// A: [M, Kp] row-major f16 (Kp = KT*32, zero-padded). Wp: packed fragments.
// Out: [M, OutStride] f16 (or f32 if OutF). grid = (M/128, ceil(N,64)/64).
__global__ __launch_bounds__(256) void gemm_f16_wmma(
    const _Float16* __restrict__ A, int Kp,
    const _Float16* __restrict__ Wp, int KT,
    _Float16* __restrict__ OutH, float* __restrict__ OutF,
    int OutStride, int Nreal,
    const float* __restrict__ gamma, const float* __restrict__ beta,
    int mode, const _Float16* __restrict__ aux) {
  const int lane = threadIdx.x & 31;
  const int wave = threadIdx.x >> 5;
  const int g16 = lane >> 4;   // half-wave group
  const int r16 = lane & 15;
  const size_t mbase = (size_t)blockIdx.x * 128 + (size_t)wave * 16;
  const int ntb = blockIdx.y * 64;

  v8f acc0 = {}, acc1 = {}, acc2 = {}, acc3 = {};
  const _Float16* arow = A + (mbase + r16) * (size_t)Kp + g16 * 8;
  const _Float16* wb = Wp + (size_t)(ntb >> 4) * KT * 512 + lane * 16;
  const size_t wstep = (size_t)KT * 512;

  for (int kt = 0; kt < KT; ++kt) {
    v8h lo = *(const v8h*)(arow + kt * 32);        // K = kt*32 + 8g .. +7
    v8h hi = *(const v8h*)(arow + kt * 32 + 16);   // K = kt*32 + 16 + 8g ..
    v16h a = __builtin_shufflevector(lo, hi, 0, 1, 2, 3, 4, 5, 6, 7,
                                             8, 9, 10, 11, 12, 13, 14, 15);
    const _Float16* wk = wb + (size_t)kt * 512;
    v16h b0 = *(const v16h*)(wk);
    v16h b1 = *(const v16h*)(wk + wstep);
    v16h b2 = *(const v16h*)(wk + 2 * wstep);
    v16h b3 = *(const v16h*)(wk + 3 * wstep);
    acc0 = __builtin_amdgcn_wmma_f32_16x16x32_f16(false, a, false, b0, (short)0, acc0, false, false);
    acc1 = __builtin_amdgcn_wmma_f32_16x16x32_f16(false, a, false, b1, (short)0, acc1, false, false);
    acc2 = __builtin_amdgcn_wmma_f32_16x16x32_f16(false, a, false, b2, (short)0, acc2, false, false);
    acc3 = __builtin_amdgcn_wmma_f32_16x16x32_f16(false, a, false, b3, (short)0, acc3, false, false);
  }

#define EPILOGUE(J, ACC)                                                        \
  {                                                                             \
    int n = ntb + (J)*16 + r16;                                                 \
    bool valid = (n < Nreal);                                                   \
    float sc = 1.0f, bs = 0.0f;                                                 \
    if (valid) {                                                                \
      if (gamma) sc = gamma[n] * BNRS;                                          \
      if (beta) bs = beta[n];                                                   \
    }                                                                           \
    if (n < OutStride) {                                                        \
      size_t rowb = (mbase + 8 * g16) * (size_t)OutStride + n;                  \
      _Pragma("unroll") for (int r = 0; r < 8; ++r) {                           \
        float v = (ACC)[r] * sc + bs;                                           \
        if (mode == EPI_SILU)      v = v * (1.0f / (1.0f + __expf(-v)));        \
        else if (mode == EPI_RELU) v = fmaxf(v, 0.0f);                          \
        else if (mode == EPI_SIG)  v = 1.0f / (1.0f + __expf(-v));              \
        else if (mode == EPI_MULSIG)                                            \
          v = (float)aux[rowb + (size_t)r * OutStride] *                        \
              (1.0f / (1.0f + __expf(-v)));                                     \
        else if (aux) v += (float)aux[rowb + (size_t)r * OutStride];            \
        if (!valid) v = 0.0f;                                                   \
        if (OutF) OutF[rowb + (size_t)r * OutStride] = v;                       \
        else      OutH[rowb + (size_t)r * OutStride] = (_Float16)v;             \
      }                                                                         \
    }                                                                           \
  }
  EPILOGUE(0, acc0)
  EPILOGUE(1, acc1)
  EPILOGUE(2, acc2)
  EPILOGUE(3, acc3)
#undef EPILOGUE
}

// ------------------------- stem: 3x3 s2 conv 1->32 + BN + SiLU -------------
__global__ __launch_bounds__(256) void stem_k(const float* __restrict__ x,
                                              const float* __restrict__ w,
                                              const float* __restrict__ g,
                                              const float* __restrict__ b,
                                              _Float16* __restrict__ out) {
  int idx = blockIdx.x * 256 + threadIdx.x;
  if (idx >= NIMG * 16 * 32) return;
  int oc = idx & 31;
  int pix = (idx >> 5) & 15;
  int n = idx >> 9;
  int oh = pix >> 2, ow = pix & 3;
  float acc = 0.0f;
#pragma unroll
  for (int ky = 0; ky < 3; ++ky) {
    int ih = oh * 2 - 1 + ky;
    if ((unsigned)ih < 8u)
#pragma unroll
      for (int kx = 0; kx < 3; ++kx) {
        int iw = ow * 2 - 1 + kx;
        if ((unsigned)iw < 8u)
          acc += x[(size_t)n * 64 + ih * 8 + iw] * w[oc * 9 + ky * 3 + kx];
      }
  }
  float v = acc * g[oc] * BNRS + b[oc];
  v = v * (1.0f / (1.0f + __expf(-v)));
  out[(size_t)idx] = (_Float16)v;  // NHWC, C=32=Kp
}

// ------------------------- depthwise conv (multi-pixel) + BN + SiLU --------
__global__ __launch_bounds__(256) void dw_k(const _Float16* __restrict__ in,
                                            const float* __restrict__ w,
                                            const float* __restrict__ g,
                                            const float* __restrict__ b,
                                            _Float16* __restrict__ out,
                                            int C, int Kp, int Hi, int Ho,
                                            int kk, int stride, size_t total) {
  size_t idx = (size_t)blockIdx.x * 256 + threadIdx.x;
  if (idx >= total) return;
  int c = (int)(idx % Kp);
  size_t t = idx / Kp;
  int wo = (int)(t % Ho); t /= Ho;
  int ho = (int)(t % Ho);
  size_t n = t / Ho;
  if (c >= C) { out[idx] = (_Float16)0.0f; return; }
  int pad = kk >> 1;
  float acc = 0.0f;
  for (int ky = 0; ky < kk; ++ky) {
    int hi = ho * stride - pad + ky;
    if ((unsigned)hi < (unsigned)Hi)
      for (int kx = 0; kx < kk; ++kx) {
        int wi = wo * stride - pad + kx;
        if ((unsigned)wi < (unsigned)Hi)
          acc += (float)in[((n * Hi + hi) * (size_t)Hi + wi) * Kp + c] *
                 w[c * kk * kk + ky * kk + kx];
      }
  }
  float v = acc * g[c] * BNRS + b[c];
  v = v * (1.0f / (1.0f + __expf(-v)));
  out[idx] = (_Float16)v;
}

// -------- depthwise at 1x1 spatial == per-channel scale; +BN+SiLU, in-place
__global__ __launch_bounds__(256) void ew_dw_silu(_Float16* __restrict__ h,
                                                  const float* __restrict__ w,
                                                  const float* __restrict__ g,
                                                  const float* __restrict__ b,
                                                  int C, int Kp, int cent,
                                                  int kk2, size_t total) {
  size_t idx = (size_t)blockIdx.x * 256 + threadIdx.x;
  if (idx >= total) return;
  int c = (int)(idx % Kp);
  if (c >= C) { h[idx] = (_Float16)0.0f; return; }
  float v = (float)h[idx] * w[c * kk2 + cent] * (g[c] * BNRS) + b[c];
  v = v * (1.0f / (1.0f + __expf(-v)));
  h[idx] = (_Float16)v;
}

// ------------------------- spatial mean pool [N,P,Kp] -> [N,Kp] ------------
__global__ __launch_bounds__(256) void pool_k(const _Float16* __restrict__ in,
                                              _Float16* __restrict__ out,
                                              int P, int Kp, int C, size_t total) {
  size_t idx = (size_t)blockIdx.x * 256 + threadIdx.x;
  if (idx >= total) return;
  int c = (int)(idx % Kp);
  size_t n = idx / Kp;
  if (c >= C) { out[idx] = (_Float16)0.0f; return; }
  float s = 0.0f;
  for (int p = 0; p < P; ++p) s += (float)in[(n * P + p) * (size_t)Kp + c];
  out[idx] = (_Float16)(s * (1.0f / (float)P));
}

// ------------------------- SE excite scale (multi-pixel) -------------------
__global__ __launch_bounds__(256) void sescale_k(_Float16* __restrict__ h,
                                                 const _Float16* __restrict__ y,
                                                 int P, int Kp, size_t total) {
  size_t idx = (size_t)blockIdx.x * 256 + threadIdx.x;
  if (idx >= total) return;
  int c = (int)(idx % Kp);
  size_t n = (idx / Kp) / P;
  h[idx] = (_Float16)((float)h[idx] * (float)y[n * (size_t)Kp + c]);
}

// ------------------------- final sensor-mean + transpose -------------------
// tmp: f32 [20480,128];  out[b,d,t] = mean_s tmp[((b*256+t)*5+s), d]
__global__ __launch_bounds__(256) void outred_k(const float* __restrict__ tmp,
                                                float* __restrict__ out) {
  int idx = blockIdx.x * 256 + threadIdx.x;
  if (idx >= 16 * 128 * 256) return;
  int t = idx & 255;
  int d = (idx >> 8) & 127;
  int b = idx >> 15;
  float s = 0.0f;
#pragma unroll
  for (int s5 = 0; s5 < 5; ++s5)
    s += tmp[(((size_t)b * 256 + t) * 5 + s5) * 128 + d];
  out[idx] = s * 0.2f;
}

// ===========================================================================
extern "C" void kernel_launch(void* const* d_in, const int* in_sizes, int n_in,
                              void* d_out, int out_size, void* d_ws, size_t ws_size,
                              hipStream_t stream) {
  (void)in_sizes; (void)n_in; (void)out_size; (void)ws_size;

  static const int CIN[16]  = {32, 16, 24, 24, 40, 40, 80, 80, 80, 112, 112, 112, 192, 192, 192, 192};
  static const int COUT[16] = {16, 24, 24, 40, 40, 80, 80, 80, 112, 112, 112, 192, 192, 192, 192, 320};
  static const int EXPD[16] = {1, 6, 6, 6, 6, 6, 6, 6, 6, 6, 6, 6, 6, 6, 6, 6};
  static const int KSZ[16]  = {3, 3, 3, 5, 5, 3, 3, 3, 5, 5, 5, 5, 5, 5, 5, 3};
  static const int STR[16]  = {1, 2, 1, 2, 1, 2, 1, 1, 1, 1, 1, 2, 1, 1, 1, 1};

  // ---- input index map: JAX tree-flatten order (sorted dict keys),
  //      'params' before 'x'; block dicts sorted: dw_b,dw_g,dw_w,
  //      [exp_b,exp_g,exp_w], proj_b,proj_g,proj_w, se_w1, se_w2.
  int ci = 0;
  int dwb[16], dwg[16], dww[16], expb[16], expg[16], expw[16];
  int prb[16], prg[16], prw[16], se1i[16], se2i[16];
  for (int i = 0; i < 16; ++i) {
    dwb[i] = ci++; dwg[i] = ci++; dww[i] = ci++;
    if (EXPD[i] != 1) { expb[i] = ci++; expg[i] = ci++; expw[i] = ci++; }
    else { expb[i] = expg[i] = expw[i] = -1; }
    prb[i] = ci++; prg[i] = ci++; prw[i] = ci++;
    se1i[i] = ci++; se2i[i] = ci++;
  }
  const int i_head_b = ci++, i_head_g = ci++, i_head_w = ci++;
  const int i_pb = ci++, i_pw = ci++;
  const int i_sb = ci++, i_sg = ci++, i_sw = ci++;
  const int i_x = ci++;

  auto F = [&](int i) { return (const float*)d_in[i]; };

  // ---- workspace carve
  char* ws = (char*)d_ws;
  _Float16* WPK  = (_Float16*)(ws);
  _Float16* bufY = (_Float16*)(ws + (size_t)32  * (1 << 20));
  _Float16* bufX = (_Float16*)(ws + (size_t)96  * (1 << 20));
  _Float16* bufZ = (_Float16*)(ws + (size_t)120 * (1 << 20));
  _Float16* bufP = (_Float16*)(ws + (size_t)144 * (1 << 20));
  _Float16* bufS1= (_Float16*)(ws + (size_t)152 * (1 << 20));
  _Float16* bufD = (_Float16*)(ws + (size_t)164 * (1 << 20));
  size_t woff = 0;  // halves

  struct PW { _Float16* p; int KT; };
  auto pack = [&](const float* W, int K, int N) -> PW {
    int KT = rup(K, 32) / 32;
    long total = (long)(rup(N, 64) / 16) * KT * 512;
    _Float16* p = WPK + woff;
    woff += (size_t)total;
    pack_w<<<dim3((unsigned)((total + 255) / 256)), dim3(256), 0, stream>>>(
        W, p, K, N, KT, total);
    return {p, KT};
  };
  auto gemm = [&](const _Float16* A, int K, size_t M, PW pw,
                  _Float16* outH, float* outF, int N,
                  const float* gm, const float* bt, int mode,
                  const _Float16* aux) {
    dim3 grid((unsigned)(M / 128), (unsigned)(rup(N, 64) / 64));
    gemm_f16_wmma<<<grid, dim3(256), 0, stream>>>(
        A, pw.KT * 32, pw.p, pw.KT, outH, outF, rup(N, 32), N, gm, bt, mode, aux);
  };
  auto nb = [](size_t total) { return dim3((unsigned)((total + 255) / 256)); };

  // ---- stem: [20480,1,8,8] -> NHWC f16 [20480,4,4,32] + BN + SiLU
  stem_k<<<nb((size_t)NIMG * 16 * 32), dim3(256), 0, stream>>>(
      F(i_x), F(i_sw), F(i_sg), F(i_sb), bufX);

  _Float16* cur = bufX;  // block input / residual source
  _Float16* nxt = bufZ;  // block output
  int Hi = 4;

  for (int bi = 0; bi < 16; ++bi) {
    const int cin = CIN[bi], cout = COUT[bi], kk = KSZ[bi], st = STR[bi];
    const int hid = cin * EXPD[bi];
    const int Ho = (st == 2) ? (Hi + 1) / 2 : Hi;
    const int Pin = Hi * Hi, Pout = Ho * Ho;
    const size_t Mi = (size_t)NIMG * Pin, Mo = (size_t)NIMG * Pout;
    const int KpH = rup(hid, 32);

    // expand 1x1 conv + BN + SiLU
    _Float16* h;
    if (EXPD[bi] != 1) {
      PW pe = pack(F(expw[bi]), cin, hid);
      gemm(cur, cin, Mi, pe, bufY, nullptr, hid, F(expg[bi]), F(expb[bi]),
           EPI_SILU, nullptr);
      h = bufY;
    } else {
      h = cur;
    }

    // depthwise + BN + SiLU
    _Float16* hd;
    if (Pin > 1) {
      size_t tot = Mo * (size_t)KpH;
      dw_k<<<nb(tot), dim3(256), 0, stream>>>(h, F(dww[bi]), F(dwg[bi]),
                                              F(dwb[bi]), bufD, hid, KpH, Hi,
                                              Ho, kk, st, tot);
      hd = bufD;
    } else {
      size_t tot = (size_t)NIMG * KpH;
      ew_dw_silu<<<nb(tot), dim3(256), 0, stream>>>(h, F(dww[bi]), F(dwg[bi]),
                                                    F(dwb[bi]), hid, KpH,
                                                    (kk * kk) / 2, kk * kk, tot);
      hd = h;
    }

    // squeeze-and-excitation
    const _Float16* seIn;
    if (Pout > 1) {
      size_t tot = (size_t)NIMG * KpH;
      pool_k<<<nb(tot), dim3(256), 0, stream>>>(hd, bufP, Pout, KpH, hid, tot);
      seIn = bufP;
    } else {
      seIn = hd;
    }
    PW p1 = pack(F(se1i[bi]), hid, hid / 4);
    gemm(seIn, hid, NIMG, p1, bufS1, nullptr, hid / 4, nullptr, nullptr,
         EPI_RELU, nullptr);
    PW p2 = pack(F(se2i[bi]), hid / 4, hid);
    if (Pout > 1) {
      gemm(bufS1, hid / 4, NIMG, p2, bufP, nullptr, hid, nullptr, nullptr,
           EPI_SIG, nullptr);
      size_t tot = Mo * (size_t)KpH;
      sescale_k<<<nb(tot), dim3(256), 0, stream>>>(hd, bufP, Pout, KpH, tot);
    } else {
      // fused: hd[m,c] = hd[m,c] * sigmoid(acc)  (in-place)
      gemm(bufS1, hid / 4, NIMG, p2, hd, nullptr, hid, nullptr, nullptr,
           EPI_MULSIG, hd);
    }

    // project 1x1 conv + BN (+ residual)
    PW pp = pack(F(prw[bi]), hid, cout);
    const bool resid = (st == 1 && cin == cout);
    gemm(hd, hid, Mo, pp, nxt, nullptr, cout, F(prg[bi]), F(prb[bi]),
         EPI_NONE, resid ? cur : nullptr);

    _Float16* tsw = cur; cur = nxt; nxt = tsw;
    Hi = Ho;
  }

  // ---- head: 320 -> 1280, BN + SiLU (1x1 spatial, pool is identity)
  PW ph = pack(F(i_head_w), 320, 1280);
  gemm(cur, 320, NIMG, ph, bufY, nullptr, 1280, F(i_head_g), F(i_head_b),
       EPI_SILU, nullptr);

  // ---- final projection 1280 -> 128 with bias, f32 output
  PW pf = pack(F(i_pw), 1280, 128);
  float* feat = (float*)nxt;  // f32 [20480,128] = 10.5 MB, fits 24MB region
  gemm(bufY, 1280, NIMG, pf, nullptr, feat, 128, nullptr, F(i_pb),
       EPI_NONE, nullptr);

  // ---- mean over sensors + transpose -> d_out [16,128,256] f32
  outred_k<<<nb((size_t)16 * 128 * 256), dim3(256), 0, stream>>>(
      feat, (float*)d_out);
}
